// GPT2Attention_48000554500775
// MI455X (gfx1250) — matrix operand order repaired and checked by reading, not compiled, not run on hardware
//
#include <hip/hip_runtime.h>

typedef unsigned short u16;
typedef __attribute__((ext_vector_type(16))) __bf16 v16bf;
typedef __attribute__((ext_vector_type(8)))  float  v8f;

union Frag {
  v16bf v;
  uint4 q[2];
};

__device__ __forceinline__ u16 f2bf(float f) {
  unsigned int u = __float_as_uint(f);
  u += 0x7FFFu + ((u >> 16) & 1u);   // round-to-nearest-even to bf16
  return (u16)(u >> 16);
}

__device__ __forceinline__ v8f vzero8() {
  v8f z = {0.f, 0.f, 0.f, 0.f, 0.f, 0.f, 0.f, 0.f};
  return z;
}

// async copy of 16 bytes global -> LDS, tracked by ASYNCcnt (CDNA5)
__device__ __forceinline__ void async_copy_b128(const u16* lds_dst, const u16* gsrc) {
  unsigned lds_off = (unsigned)(size_t)(lds_dst);       // flat LDS addr -> 32-bit offset
  unsigned long long ga = (unsigned long long)(size_t)gsrc;
  asm volatile("global_load_async_to_lds_b128 %0, %1, off"
               :: "v"(lds_off), "v"(ga) : "memory");
}
__device__ __forceinline__ void wait_async0() {
  asm volatile("s_wait_asynccnt 0" ::: "memory");
}

// ---------------- conversion kernels ----------------

__global__ void __launch_bounds__(256) k_f32_to_bf16_v4(const float4* __restrict__ in,
                                                        uint2* __restrict__ out, int n4) {
  int i = blockIdx.x * blockDim.x + threadIdx.x;
  if (i < n4) {
    float4 v = in[i];
    uint2 r;
    r.x = (unsigned)f2bf(v.x) | ((unsigned)f2bf(v.y) << 16);
    r.y = (unsigned)f2bf(v.z) | ((unsigned)f2bf(v.w) << 16);
    out[i] = r;
  }
}

// W[k*N + n] (f32, [K][N]) -> Wt[n*K + k] (bf16, [N][K])
__global__ void __launch_bounds__(256) k_transpose_bf16(const float* __restrict__ W,
                                                        u16* __restrict__ Wt, int K, int N) {
  int i = blockIdx.x * blockDim.x + threadIdx.x;
  if (i >= K * N) return;
  int n = i / K;
  int k = i - n * K;
  Wt[(size_t)n * K + k] = f2bf(W[(size_t)k * N + n]);
}

// ---------------- async-LDS double-buffered GEMM core ----------------
// Block (8 waves) computes 128x64 tile.  A: [M][K] bf16, Bt: [N][K] bf16.
// LDS rows padded to 40 elements (80B) -> conflict-free b128 fragment reads.
#define LROW 40

template <int K>
__device__ __forceinline__ void gemm_block_128x64(const u16* __restrict__ A,
                                                  const u16* __restrict__ Bt,
                                                  int m0, int n0, int tid,
                                                  u16* shA, u16* shB,   // [2][128*LROW], [2][64*LROW]
                                                  v8f acc[4]) {
  const int lane = tid & 31;
  const int wid  = tid >> 5;
  const int row   = lane & 15;
  const int koffA = (lane < 16) ? 0 : 8;
  const int koffB = (lane < 16) ? 0 : 8;

  // copy assignments: A = 512 b128 chunks (2/thread), B = 256 chunks (1/thread)
  const int ar0 = tid >> 2,         ap0 = tid & 3;          // chunk tid
  const int ar1 = (tid + 256) >> 2, ap1 = (tid + 256) & 3;  // chunk tid+256
  const int br  = tid >> 2,         bp  = tid & 3;

  auto stage = [&](int bb, int k0) {
    async_copy_b128(shA + bb * 128 * LROW + ar0 * LROW + ap0 * 8,
                    A + (size_t)(m0 + ar0) * K + k0 + ap0 * 8);
    async_copy_b128(shA + bb * 128 * LROW + ar1 * LROW + ap1 * 8,
                    A + (size_t)(m0 + ar1) * K + k0 + ap1 * 8);
    async_copy_b128(shB + bb * 64 * LROW + br * LROW + bp * 8,
                    Bt + (size_t)(n0 + br) * K + k0 + bp * 8);
  };

  auto compute = [&](int bb) {
    const u16* pa = shA + bb * 128 * LROW + (wid * 16 + row) * LROW;
    Frag a;
    a.q[0] = *(const uint4*)(pa + koffA);
    a.q[1] = *(const uint4*)(pa + 16 + koffA);
#pragma unroll
    for (int t = 0; t < 4; ++t) {
      const u16* pb = shB + bb * 64 * LROW + (t * 16 + row) * LROW + koffB;
      Frag b;
      b.q[0] = *(const uint4*)(pb);
      b.q[1] = *(const uint4*)(pb + 16);
      acc[t] = __builtin_amdgcn_wmma_f32_16x16x32_bf16(false, a.v, false, b.v,
                                                       (short)0, acc[t], false, false);
    }
  };

  constexpr int NSTEP = K / 32;
  stage(0, 0);
  wait_async0();
  __syncthreads();
#pragma unroll 2
  for (int i = 0; i < NSTEP; ++i) {
    const int cur = i & 1;
    if (i + 1 < NSTEP) stage(cur ^ 1, (i + 1) * 32);  // copies in flight during WMMAs
    compute(cur);
    wait_async0();                                    // own copies for next buf landed
    __syncthreads();                                  // everyone done reading cur buf
  }
}

// ---------------- QKV GEMM with fused reshape/scatter ----------------
// out layout: Qb,Kb = [B,H,T,D] bf16 ; Vt = [B,H,D,T] bf16
__global__ void __launch_bounds__(256) k_gemm_qkv(const u16* __restrict__ xb,
                                                  const u16* __restrict__ Wt,
                                                  const float* __restrict__ bias,
                                                  u16* __restrict__ Qb,
                                                  u16* __restrict__ Kb,
                                                  u16* __restrict__ Vt) {
  __shared__ u16 shA[2][128 * LROW];
  __shared__ u16 shB[2][64 * LROW];
  const int C = 1024, N = 3072, T = 2048;
  const int tid = threadIdx.x, wid = tid >> 5, lane = tid & 31;
  const int gn = N / 64;                              // 48
  const int bm = blockIdx.x / gn, bn = blockIdx.x - bm * gn;
  const int m0 = bm * 128, n0 = bn * 64;

  v8f acc[4];
#pragma unroll
  for (int t = 0; t < 4; ++t) acc[t] = vzero8();

  gemm_block_128x64<1024>(xb, Wt, m0, n0, tid, &shA[0][0], &shB[0][0], acc);

  const int m0w = m0 + wid * 16;                      // this wave's 16 rows
  const int s = n0 / C;                               // 0=q 1=k 2=v
  const int h = (n0 % C) / 64;
  const int b = m0w / T;
  const int tt0 = m0w % T;
  const int rbase = (lane >> 4) << 3;
  const int colb  = lane & 15;
  const size_t bh = (size_t)b * 16 + h;

#pragma unroll
  for (int t = 0; t < 4; ++t) {
    const int d = t * 16 + colb;
    const float bi = bias[n0 + d];
#pragma unroll
    for (int r = 0; r < 8; ++r) {
      const int tt = tt0 + rbase + r;
      const u16 val = f2bf(acc[t][r] + bi);
      if (s == 0)      Qb[(bh * 2048 + tt) * 64 + d] = val;
      else if (s == 1) Kb[(bh * 2048 + tt) * 64 + d] = val;
      else             Vt[(bh * 64 + d) * 2048 + tt] = val;
    }
  }
}

// ---------------- flash attention (one wave = 16 queries of one head) ----------------
__global__ void __launch_bounds__(256) k_attn(const u16* __restrict__ Qb,
                                              const u16* __restrict__ Kb,
                                              const u16* __restrict__ Vt,
                                              u16* __restrict__ Yb) {
  __shared__ u16 plds[8][16][32];                 // per-wave P staging (bf16)
  const int T = 2048, D = 64, C = 1024;
  const int wid = threadIdx.x >> 5, lane = threadIdx.x & 31;
  const int tile = blockIdx.x * 8 + wid;          // 8192 tiles
  const int bh = tile >> 7;
  const int q0 = (tile & 127) << 4;

  const u16* Qp = Qb + (size_t)bh * T * D;
  const u16* Kp = Kb + (size_t)bh * T * D;
  const u16* Vp = Vt + (size_t)bh * D * T;

  const int row   = lane & 15;
  const int koffA = (lane < 16) ? 0 : 8;
  const int koffB = (lane < 16) ? 0 : 8;
  const int rbase = (lane >> 4) << 3;

  Frag qa[2];
#pragma unroll
  for (int j = 0; j < 2; ++j) {
    const u16* qp = Qp + (size_t)(q0 + row) * D + j * 32 + koffA;
    qa[j].q[0] = *(const uint4*)(qp);
    qa[j].q[1] = *(const uint4*)(qp + 16);
  }

  v8f o[4];
  float m[8], l[8];
#pragma unroll
  for (int t = 0; t < 4; ++t) o[t] = vzero8();
#pragma unroll
  for (int r = 0; r < 8; ++r) { m[r] = -3.0e38f; l[r] = 0.f; }

  const float scale = 0.125f;  // 1/sqrt(64)
  const int kend = q0 + 16;

  auto load_kblock = [&](Frag (&kb)[2][2], int k0) {
#pragma unroll
    for (int u = 0; u < 2; ++u)
#pragma unroll
      for (int j = 0; j < 2; ++j) {
        const u16* kp = Kp + (size_t)(k0 + u * 16 + row) * D + j * 32 + koffB;
        kb[u][j].q[0] = *(const uint4*)(kp);
        kb[u][j].q[1] = *(const uint4*)(kp + 16);
      }
  };

  auto step = [&](Frag (&kcur)[2][2], Frag (&knxt)[2][2], int k0) {
    Frag vb[4];                                   // V loads early; waited after softmax
#pragma unroll
    for (int t = 0; t < 4; ++t) {
      const u16* vp = Vp + (size_t)(t * 16 + row) * T + k0 + koffB;
      vb[t].q[0] = *(const uint4*)(vp);
      vb[t].q[1] = *(const uint4*)(vp + 16);
    }
    if (k0 + 32 < kend) load_kblock(knxt, k0 + 32);

    v8f s0t = vzero8(), s1t = vzero8();
#pragma unroll
    for (int j = 0; j < 2; ++j) {
      s0t = __builtin_amdgcn_wmma_f32_16x16x32_bf16(false, qa[j].v, false, kcur[0][j].v,
                                                    (short)0, s0t, false, false);
      s1t = __builtin_amdgcn_wmma_f32_16x16x32_bf16(false, qa[j].v, false, kcur[1][j].v,
                                                    (short)0, s1t, false, false);
    }

    const bool needmask = (k0 + 31 > q0);
#pragma unroll
    for (int r = 0; r < 8; ++r) {
      const int qq = q0 + rbase + r;
      float v0 = s0t[r] * scale;
      float v1 = s1t[r] * scale;
      if (needmask) {
        if (k0 + row > qq)      v0 = -3.0e38f;
        if (k0 + 16 + row > qq) v1 = -3.0e38f;
      }
      float mx = fmaxf(v0, v1);
#pragma unroll
      for (int off = 1; off < 16; off <<= 1)
        mx = fmaxf(mx, __shfl_xor(mx, off, 32));
      const float mnew  = fmaxf(m[r], mx);
      const float alpha = __expf(m[r] - mnew);
      const float p0 = __expf(v0 - mnew);
      const float p1 = __expf(v1 - mnew);
      float rs = p0 + p1;
#pragma unroll
      for (int off = 1; off < 16; off <<= 1)
        rs += __shfl_xor(rs, off, 32);
      l[r] = l[r] * alpha + rs;
      m[r] = mnew;
#pragma unroll
      for (int t = 0; t < 4; ++t) o[t][r] *= alpha;
      plds[wid][rbase + r][row]      = f2bf(p0);
      plds[wid][rbase + r][16 + row] = f2bf(p1);
    }
    asm volatile("s_wait_dscnt 0" ::: "memory");
    Frag pa;
    pa.q[0] = *(const uint4*)&plds[wid][row][koffA];
    pa.q[1] = *(const uint4*)&plds[wid][row][16 + koffA];
#pragma unroll
    for (int t = 0; t < 4; ++t)
      o[t] = __builtin_amdgcn_wmma_f32_16x16x32_bf16(false, pa.v, false, vb[t].v,
                                                     (short)0, o[t], false, false);
  };

  Frag kb[2][2][2];
  load_kblock(kb[0], 0);
  int k0 = 0;
  for (;;) {
    step(kb[0], kb[1], k0);
    k0 += 32; if (k0 >= kend) break;
    step(kb[1], kb[0], k0);
    k0 += 32; if (k0 >= kend) break;
  }

  float inv[8];
#pragma unroll
  for (int r = 0; r < 8; ++r) inv[r] = 1.0f / l[r];

  const int b = bh >> 4, h = bh & 15;
#pragma unroll
  for (int t = 0; t < 4; ++t) {
    const int col = h * 64 + t * 16 + (lane & 15);
#pragma unroll
    for (int r = 0; r < 8; ++r) {
      const int tt = q0 + rbase + r;
      Yb[((size_t)b * T + tt) * C + col] = f2bf(o[t][r] * inv[r]);
    }
  }
}

// ---------------- output projection GEMM ----------------
__global__ void __launch_bounds__(256) k_gemm_proj(const u16* __restrict__ Yb,
                                                   const u16* __restrict__ Wt,
                                                   const float* __restrict__ bias,
                                                   float* __restrict__ out) {
  __shared__ u16 shA[2][128 * LROW];
  __shared__ u16 shB[2][64 * LROW];
  const int C = 1024;
  const int tid = threadIdx.x, wid = tid >> 5, lane = tid & 31;
  const int gn = C / 64;                              // 16
  const int bm = blockIdx.x / gn, bn = blockIdx.x - bm * gn;
  const int m0 = bm * 128, n0 = bn * 64;

  v8f acc[4];
#pragma unroll
  for (int t = 0; t < 4; ++t) acc[t] = vzero8();

  gemm_block_128x64<1024>(Yb, Wt, m0, n0, tid, &shA[0][0], &shB[0][0], acc);

  const int m0w = m0 + wid * 16;
  const int rbase = (lane >> 4) << 3;
  const int colb  = lane & 15;
#pragma unroll
  for (int t = 0; t < 4; ++t) {
    const int n = n0 + t * 16 + colb;
    const float bi = bias[n];
#pragma unroll
    for (int r = 0; r < 8; ++r)
      out[(size_t)(m0w + rbase + r) * C + n] = acc[t][r] + bi;
  }
}

extern "C" void kernel_launch(void* const* d_in, const int* in_sizes, int n_in,
                              void* d_out, int out_size, void* d_ws, size_t ws_size,
                              hipStream_t stream) {
  (void)in_sizes; (void)n_in; (void)out_size; (void)ws_size;
  const float* x     = (const float*)d_in[0];
  const float* Wqkv  = (const float*)d_in[1];
  const float* bqkv  = (const float*)d_in[2];
  const float* Wproj = (const float*)d_in[3];
  const float* bproj = (const float*)d_in[4];
  float* out = (float*)d_out;

  const int B = 4, T = 2048, C = 1024;
  const size_t MT = (size_t)B * T;               // 8192 rows

  // workspace carve-out (all bf16): 88 MB total
  char* p = (char*)d_ws;
  u16* xb     = (u16*)p;  p += MT * C * 2;            // 16 MB
  u16* wqkvt  = (u16*)p;  p += (size_t)3 * C * C * 2; // 6 MB  [3C][C]
  u16* wprojt = (u16*)p;  p += (size_t)C * C * 2;     // 2 MB  [C][C]
  u16* Qb     = (u16*)p;  p += MT * C * 2;            // 16 MB [B,H,T,D]
  u16* Kb     = (u16*)p;  p += MT * C * 2;            // 16 MB [B,H,T,D]
  u16* Vt     = (u16*)p;  p += MT * C * 2;            // 16 MB [B,H,D,T]
  u16* Yb     = (u16*)p;                              // 16 MB [B,T,C]

  const int n4 = (int)((MT * C) / 4);
  k_f32_to_bf16_v4<<<n4 / 256, 256, 0, stream>>>((const float4*)x, (uint2*)xb, n4);
  k_transpose_bf16<<<(3 * C * C) / 256, 256, 0, stream>>>(Wqkv, wqkvt, C, 3 * C);
  k_transpose_bf16<<<(C * C) / 256, 256, 0, stream>>>(Wproj, wprojt, C, C);
  k_gemm_qkv<<<64 * 48, 256, 0, stream>>>(xb, wqkvt, bqkv, Qb, Kb, Vt);   // 128x64 tiles
  k_attn<<<1024, 256, 0, stream>>>(Qb, Kb, Vt, Yb);
  k_gemm_proj<<<64 * 16, 256, 0, stream>>>(Yb, wprojt, bproj, out);
}
